// SpatialCrossAttention_34617436406078
// MI455X (gfx1250) — compile-verified, matrix operand order checked by reading, not run on hardware
//
#include <hip/hip_runtime.h>
#include <hip/hip_fp16.h>

typedef __attribute__((ext_vector_type(16))) _Float16 v16h;
typedef __attribute__((ext_vector_type(8)))  float    v8f;

#define B_    32
#define CQ_   256
#define CC_   3
#define COUT_ 256
#define H_    32
#define W_    32
#define HW_   1024
#define INTER_ 128
#define HC_   224
#define WC_   224
#define EPS_  1e-5f

#define WMMA_F16(a, b, c) \
  __builtin_amdgcn_wmma_f32_16x16x32_f16(false, (a), false, (b), (short)0, (c), false, false)

// ---------------------------------------------------------------------------
// K0a: query [B,CQ,HW] f32 -> qT_h [B,HW,CQ] f16 (row per pixel => WMMA A rows)
// ---------------------------------------------------------------------------
__global__ void k_transpose_query(const float* __restrict__ q, _Float16* __restrict__ qT) {
  int idx = blockIdx.x * blockDim.x + threadIdx.x;      // over B*HW
  if (idx >= B_ * HW_) return;
  int b = idx >> 10, i = idx & 1023;
  const float* src = q + (size_t)b * CQ_ * HW_ + i;
  _Float16* dst = qT + (size_t)idx * CQ_;
#pragma unroll 8
  for (int c = 0; c < CQ_; ++c) dst[c] = (_Float16)src[(size_t)c * HW_];
}

// ---------------------------------------------------------------------------
// K0b: Wq [128,256], Wp [256,256] f32 -> f16 (row-major: lane-n B-operand rows)
// ---------------------------------------------------------------------------
__global__ void k_convert_weights(const float* __restrict__ Wq, const float* __restrict__ Wp,
                                  _Float16* __restrict__ Wq_h, _Float16* __restrict__ Wp_h) {
  int idx = blockIdx.x * blockDim.x + threadIdx.x;
  if (idx < INTER_ * CQ_) Wq_h[idx] = (_Float16)Wq[idx];
  if (idx < COUT_ * CQ_)  Wp_h[idx] = (_Float16)Wp[idx];
}

// ---------------------------------------------------------------------------
// K1: bilinear resize (224->32, align_corners=False) + k/v 1x1 conv (Cin=3).
//     k_h  [B,HW,INTER] f16   (j-major rows   => B-operand for scores GEMM)
//     vT_h [B,CQ,HW]    f16   (c-major rows   => B-operand for P*V^T GEMM)
// ---------------------------------------------------------------------------
__global__ void k_resize_kv(const float* __restrict__ ctx,
                            const float* __restrict__ Wk, const float* __restrict__ bk,
                            const float* __restrict__ Wv, const float* __restrict__ bv,
                            _Float16* __restrict__ k_h, _Float16* __restrict__ vT_h) {
  int idx = blockIdx.x * blockDim.x + threadIdx.x;      // over B*HW
  if (idx >= B_ * HW_) return;
  int b = idx >> 10, j = idx & 1023;
  int y = j >> 5, x = j & 31;
  float sy = (y + 0.5f) * ((float)HC_ / (float)H_) - 0.5f;
  float sx = (x + 0.5f) * ((float)WC_ / (float)W_) - 0.5f;
  sy = fmaxf(sy, 0.0f); sx = fmaxf(sx, 0.0f);
  int y0 = (int)sy, x0 = (int)sx;
  int y1 = min(y0 + 1, HC_ - 1), x1 = min(x0 + 1, WC_ - 1);
  float fy = sy - (float)y0, fx = sx - (float)x0;

  float c[CC_];
  const float* base = ctx + (size_t)b * CC_ * HC_ * WC_;
#pragma unroll
  for (int ch = 0; ch < CC_; ++ch) {
    const float* p = base + (size_t)ch * HC_ * WC_;
    float v00 = p[y0 * WC_ + x0], v01 = p[y0 * WC_ + x1];
    float v10 = p[y1 * WC_ + x0], v11 = p[y1 * WC_ + x1];
    c[ch] = (1.0f - fy) * ((1.0f - fx) * v00 + fx * v01) +
            fy          * ((1.0f - fx) * v10 + fx * v11);
  }
  _Float16* kout = k_h + (size_t)idx * INTER_;
#pragma unroll 4
  for (int ci = 0; ci < INTER_; ++ci)
    kout[ci] = (_Float16)(Wk[ci * CC_ + 0] * c[0] + Wk[ci * CC_ + 1] * c[1] +
                          Wk[ci * CC_ + 2] * c[2] + bk[ci]);
#pragma unroll 4
  for (int cc = 0; cc < CQ_; ++cc)
    vT_h[((size_t)b * CQ_ + cc) * HW_ + j] =
        (_Float16)(Wv[cc * CC_ + 0] * c[0] + Wv[cc * CC_ + 1] * c[1] +
                   Wv[cc * CC_ + 2] * c[2] + bv[cc]);
}

// ---------------------------------------------------------------------------
// K2: q projection. One wave per 16x16 output tile, K=256 -> 8 WMMAs.
//     q_h [B,HW,INTER] f16
// ---------------------------------------------------------------------------
__global__ __launch_bounds__(32) void k_qproj(const _Float16* __restrict__ qT,
                                              const _Float16* __restrict__ Wq_h,
                                              const float* __restrict__ bq,
                                              _Float16* __restrict__ q_h) {
  int tile = blockIdx.x;                 // B * 64 * 8
  int nt = tile & 7;  tile >>= 3;
  int it = tile & 63; int b = tile >> 6;
  int lane = threadIdx.x, half = lane >> 4, lid = lane & 15;
  int i0 = it * 16, n0 = nt * 16;

  const _Float16* arow = qT   + ((size_t)b * HW_ + i0 + lid) * CQ_ + 16 * half;
  const _Float16* brow = Wq_h + (size_t)(n0 + lid) * CQ_ + 16 * half;
  v8f acc = {};
#pragma unroll
  for (int k0 = 0; k0 < CQ_; k0 += 32) {
    v16h a  = *(const v16h*)(arow + k0);
    v16h bm = *(const v16h*)(brow + k0);
    acc = WMMA_F16(a, bm, acc);
  }
  float bias = bq[n0 + lid];
  _Float16* dst = q_h + ((size_t)b * HW_ + i0 + 8 * half) * INTER_ + n0 + lid;
#pragma unroll
  for (int r = 0; r < 8; ++r)
    dst[(size_t)r * INTER_] = (_Float16)(acc[r] + bias);
}

// ---------------------------------------------------------------------------
// K3: softmax statistics. One wave per (b, 16-row i-tile). A tile (q) is
//     hoisted into 32 VGPRs; only K rows stream through the j loop.
// ---------------------------------------------------------------------------
__global__ __launch_bounds__(32) void k_softmax_stats(const _Float16* __restrict__ q_h,
                                                      const _Float16* __restrict__ k_h,
                                                      float* __restrict__ m_ws,
                                                      float* __restrict__ linv_ws) {
  int tile = blockIdx.x;                 // B * 64
  int it = tile & 63; int b = tile >> 6;
  int lane = threadIdx.x, half = lane >> 4, lid = lane & 15;
  int i0 = it * 16;

  const _Float16* arow = q_h + ((size_t)b * HW_ + i0 + lid) * INTER_ + 16 * half;
  v16h aQ[4];
#pragma unroll
  for (int t = 0; t < 4; ++t) aQ[t] = *(const v16h*)(arow + t * 32);

  float mx[8], sm[8];
#pragma unroll
  for (int r = 0; r < 8; ++r) { mx[r] = -__builtin_huge_valf(); sm[r] = 0.0f; }

  for (int j0 = 0; j0 < HW_; j0 += 16) {
    const _Float16* brow = k_h + ((size_t)b * HW_ + j0 + lid) * INTER_ + 16 * half;
    v8f s = {};
#pragma unroll
    for (int t = 0; t < 4; ++t) {
      v16h bm = *(const v16h*)(brow + t * 32);
      s = WMMA_F16(aQ[t], bm, s);
    }
#pragma unroll
    for (int r = 0; r < 8; ++r) {
      float val = s[r];
      float mn  = fmaxf(mx[r], val);
      sm[r] = sm[r] * __expf(mx[r] - mn) + __expf(val - mn);
      mx[r] = mn;
    }
  }
  // combine the 16 lanes of each half-wave group (each row lives in one group)
#pragma unroll
  for (int off = 1; off < 16; off <<= 1) {
#pragma unroll
    for (int r = 0; r < 8; ++r) {
      float om = __shfl_xor(mx[r], off, 32);
      float os = __shfl_xor(sm[r], off, 32);
      float mn = fmaxf(mx[r], om);
      sm[r] = sm[r] * __expf(mx[r] - mn) + os * __expf(om - mn);
      mx[r] = mn;
    }
  }
  if (lid == 0) {
#pragma unroll
    for (int r = 0; r < 8; ++r) {
      int row = i0 + r + 8 * half;
      m_ws[b * HW_ + row]    = mx[r];
      linv_ws[b * HW_ + row] = 1.0f / sm[r];
    }
  }
}

// ---------------------------------------------------------------------------
// Async-stage one contiguous 8 KB K-tile (32 j-rows x 128 ci, f16) into LDS.
// 16 x global_load_async_to_lds_b128: each moves 32 lanes x 16 B = 512 B.
// Tracked by ASYNCcnt.
// ---------------------------------------------------------------------------
__device__ __forceinline__ void stage_k_tile(const _Float16* gsrc, _Float16* lds_dst,
                                             int lane) {
  unsigned lds = (unsigned)(size_t)lds_dst + (unsigned)lane * 16u;
  unsigned long long g = (unsigned long long)(size_t)gsrc + (unsigned long long)lane * 16ull;
#pragma unroll
  for (int c = 0; c < 16; ++c) {
    asm volatile("global_load_async_to_lds_b128 %0, %1, off"
                 :: "v"(lds + (unsigned)(c * 512)),
                    "v"(g + (unsigned long long)(c * 512))
                 : "memory");
  }
}

// ---------------------------------------------------------------------------
// K4: attended = softmax(S) @ V^T, flash-style. One wave owns 16 i x 128 c.
//     q A-tile hoisted to VGPRs; K tiles double-buffered in LDS via
//     global_load_async_to_lds_b128 + s_wait_asynccnt; P repacked via LDS.
// ---------------------------------------------------------------------------
__global__ __launch_bounds__(32) void k_attend(const _Float16* __restrict__ q_h,
                                               const _Float16* __restrict__ k_h,
                                               const _Float16* __restrict__ vT_h,
                                               const float* __restrict__ m_ws,
                                               const float* __restrict__ linv_ws,
                                               _Float16* __restrict__ att_h) {
  __shared__ __align__(64) _Float16 kbuf[2][32 * INTER_];   // 2 x 8 KB double buffer
  __shared__ __align__(64) _Float16 pT[16 * 32];            // P tile, row-major [i][j]
  int tile = blockIdx.x;                 // B * 64 * 2
  int chnk = tile & 1; tile >>= 1;
  int it = tile & 63; int b = tile >> 6;
  int lane = threadIdx.x, half = lane >> 4, lid = lane & 15;
  int i0 = it * 16, cbase = chnk * 128;

  const _Float16* arow = q_h + ((size_t)b * HW_ + i0 + lid) * INTER_ + 16 * half;
  v16h aQ[4];
#pragma unroll
  for (int t = 0; t < 4; ++t) aQ[t] = *(const v16h*)(arow + t * 32);

  float mrow[8];
#pragma unroll
  for (int r = 0; r < 8; ++r) mrow[r] = m_ws[b * HW_ + i0 + r + 8 * half];

  v8f O[8];
#pragma unroll
  for (int t = 0; t < 8; ++t) O[t] = (v8f){};

  const _Float16* kgbase = k_h + (size_t)b * HW_ * INTER_;
  stage_k_tile(kgbase, kbuf[0], lane);                      // preload block 0
  int buf = 0;

  for (int jb = 0; jb < HW_; jb += 32) {
    if (jb + 32 < HW_) {
      stage_k_tile(kgbase + (size_t)(jb + 32) * INTER_, kbuf[buf ^ 1], lane);
      asm volatile("s_wait_asynccnt 0x10" ::: "memory");    // current buffer landed
    } else {
      asm volatile("s_wait_asynccnt 0x0" ::: "memory");
    }
    // warm GL2 for a slice of the next V block
    if (jb + 32 < HW_)
      __builtin_prefetch(vT_h + ((size_t)b * CQ_ + cbase + lane) * HW_ + jb + 32, 0, 0);

#pragma unroll
    for (int sj = 0; sj < 2; ++sj) {
      const _Float16* brow = kbuf[buf] + (sj * 16 + lid) * INTER_ + 16 * half;
      v8f s = {};
#pragma unroll
      for (int t = 0; t < 4; ++t) {
        v16h bm = *(const v16h*)(brow + t * 32);             // ds_load_b128
        s = WMMA_F16(aQ[t], bm, s);
      }
#pragma unroll
      for (int r = 0; r < 8; ++r) {
        float p = __expf(s[r] - mrow[r]);
        pT[(r + 8 * half) * 32 + sj * 16 + lid] = (_Float16)p;  // ds_store
      }
    }
    __syncthreads();                                        // order pT store -> load
    v16h pa = *(const v16h*)(pT + lid * 32 + 16 * half);    // ds_load, A-layout
#pragma unroll
    for (int t = 0; t < 8; ++t) {
      const _Float16* vrow =
          vT_h + ((size_t)b * CQ_ + cbase + t * 16 + lid) * HW_ + jb + 16 * half;
      v16h bm = *(const v16h*)vrow;
      O[t] = WMMA_F16(pa, bm, O[t]);
    }
    __syncthreads();
    buf ^= 1;
  }

  float li[8];
#pragma unroll
  for (int r = 0; r < 8; ++r) li[r] = linv_ws[b * HW_ + i0 + r + 8 * half];
#pragma unroll
  for (int t = 0; t < 8; ++t) {
    int c = cbase + t * 16 + lid;
    _Float16* dst = att_h + ((size_t)b * HW_ + i0 + 8 * half) * CQ_ + c;
#pragma unroll
    for (int r = 0; r < 8; ++r)
      dst[(size_t)r * CQ_] = (_Float16)(O[t][r] * li[r]);
  }
}

// ---------------------------------------------------------------------------
// K5a: zero the per-batch GroupNorm accumulators
// ---------------------------------------------------------------------------
__global__ void k_zero_sums(float* __restrict__ sums) {
  int idx = blockIdx.x * blockDim.x + threadIdx.x;
  if (idx < B_ * 2) sums[idx] = 0.0f;
}

// ---------------------------------------------------------------------------
// K5: fused = Wp @ (query + attended) + bp, written to d_out [B,COUT,HW],
//     plus wave-reduced atomic sum/sumsq per batch for GroupNorm.
// ---------------------------------------------------------------------------
__global__ __launch_bounds__(32) void k_fuse_proj(const _Float16* __restrict__ qT_h,
                                                  const _Float16* __restrict__ att_h,
                                                  const _Float16* __restrict__ Wp_h,
                                                  const float* __restrict__ bp,
                                                  float* __restrict__ out,
                                                  float* __restrict__ sums) {
  int tile = blockIdx.x;                 // B * 64 * 16
  int nt = tile & 15; tile >>= 4;
  int it = tile & 63; int b = tile >> 6;
  int lane = threadIdx.x, half = lane >> 4, lid = lane & 15;
  int i0 = it * 16, n0 = nt * 16;

  const _Float16* xq = qT_h  + ((size_t)b * HW_ + i0 + lid) * CQ_ + 16 * half;
  const _Float16* xa = att_h + ((size_t)b * HW_ + i0 + lid) * CQ_ + 16 * half;
  const _Float16* brow = Wp_h + (size_t)(n0 + lid) * CQ_ + 16 * half;
  v8f acc = {};
#pragma unroll
  for (int k0 = 0; k0 < CQ_; k0 += 32) {
    v16h a  = *(const v16h*)(xq + k0) + *(const v16h*)(xa + k0);  // v_pk_add_f16
    v16h bm = *(const v16h*)(brow + k0);
    acc = WMMA_F16(a, bm, acc);
  }
  float bias = bp[n0 + lid];
  float s = 0.0f, s2 = 0.0f;
  float* dst = out + ((size_t)b * COUT_ + n0 + lid) * HW_ + i0 + 8 * half;
#pragma unroll
  for (int r = 0; r < 8; ++r) {
    float v = acc[r] + bias;
    dst[r] = v;
    s += v; s2 += v * v;
  }
#pragma unroll
  for (int off = 1; off < 32; off <<= 1) {
    s  += __shfl_xor(s,  off, 32);
    s2 += __shfl_xor(s2, off, 32);
  }
  if (lane == 0) {
    atomicAdd(&sums[b * 2 + 0], s);
    atomicAdd(&sums[b * 2 + 1], s2);
  }
}

// ---------------------------------------------------------------------------
// K6: GroupNorm(1 group) normalize in place + affine
// ---------------------------------------------------------------------------
__global__ void k_groupnorm(float* __restrict__ out, const float* __restrict__ sums,
                            const float* __restrict__ gamma, const float* __restrict__ beta) {
  int idx = blockIdx.x * blockDim.x + threadIdx.x;
  if (idx >= B_ * COUT_ * HW_) return;
  int b = idx >> 18;                     // / (COUT_*HW_) = 262144
  int c = (idx >> 10) & (COUT_ - 1);
  const float invN = 1.0f / (float)(COUT_ * HW_);
  float mu  = sums[b * 2 + 0] * invN;
  float var = sums[b * 2 + 1] * invN - mu * mu;
  float x = out[idx];
  out[idx] = (x - mu) * rsqrtf(var + EPS_) * gamma[c] + beta[c];
}

// ---------------------------------------------------------------------------
extern "C" void kernel_launch(void* const* d_in, const int* in_sizes, int n_in,
                              void* d_out, int out_size, void* d_ws, size_t ws_size,
                              hipStream_t stream) {
  const float* query = (const float*)d_in[0];
  const float* ctx   = (const float*)d_in[1];
  const float* Wq    = (const float*)d_in[2];
  const float* bq    = (const float*)d_in[3];
  const float* Wk    = (const float*)d_in[4];
  const float* bk    = (const float*)d_in[5];
  const float* Wv    = (const float*)d_in[6];
  const float* bv    = (const float*)d_in[7];
  const float* Wp    = (const float*)d_in[8];
  const float* bp    = (const float*)d_in[9];
  const float* gamma = (const float*)d_in[10];
  const float* beta  = (const float*)d_in[11];
  float* out = (float*)d_out;

  // ---- carve workspace (256B aligned slabs) ----
  char* ws = (char*)d_ws;
  size_t off = 0;
  auto carve = [&](size_t bytes) {
    char* p = ws + off;
    off += (bytes + 255) & ~(size_t)255;
    return p;
  };
  _Float16* qT_h   = (_Float16*)carve((size_t)B_ * HW_ * CQ_ * 2);     // 16 MB
  _Float16* Wq_h   = (_Float16*)carve((size_t)INTER_ * CQ_ * 2);
  _Float16* Wp_h   = (_Float16*)carve((size_t)COUT_ * CQ_ * 2);
  _Float16* k_h    = (_Float16*)carve((size_t)B_ * HW_ * INTER_ * 2);  // 8 MB
  _Float16* vT_h   = (_Float16*)carve((size_t)B_ * CQ_ * HW_ * 2);     // 16 MB
  _Float16* q_h    = (_Float16*)carve((size_t)B_ * HW_ * INTER_ * 2);  // 8 MB
  float*    m_ws   = (float*)carve((size_t)B_ * HW_ * 4);
  float*    linv_ws= (float*)carve((size_t)B_ * HW_ * 4);
  _Float16* att_h  = (_Float16*)carve((size_t)B_ * HW_ * CQ_ * 2);     // 16 MB
  float*    sums   = (float*)carve((size_t)B_ * 2 * 4);
  (void)ws_size; (void)in_sizes; (void)n_in; (void)out_size;

  // ---- pipeline ----
  k_transpose_query<<<(B_ * HW_ + 255) / 256, 256, 0, stream>>>(query, qT_h);
  k_convert_weights<<<(COUT_ * CQ_ + 255) / 256, 256, 0, stream>>>(Wq, Wp, Wq_h, Wp_h);
  k_resize_kv<<<(B_ * HW_ + 255) / 256, 256, 0, stream>>>(ctx, Wk, bk, Wv, bv, k_h, vT_h);
  k_qproj<<<B_ * (HW_ / 16) * (INTER_ / 16), 32, 0, stream>>>(qT_h, Wq_h, bq, q_h);
  k_softmax_stats<<<B_ * (HW_ / 16), 32, 0, stream>>>(q_h, k_h, m_ws, linv_ws);
  k_attend<<<B_ * (HW_ / 16) * 2, 32, 0, stream>>>(q_h, k_h, vT_h, m_ws, linv_ws, att_h);
  k_zero_sums<<<1, 64, 0, stream>>>(sums);
  k_fuse_proj<<<B_ * (HW_ / 16) * (COUT_ / 16), 32, 0, stream>>>(qT_h, att_h, Wp_h, bp, out, sums);
  k_groupnorm<<<(B_ * COUT_ * HW_ + 255) / 256, 256, 0, stream>>>(out, sums, gamma, beta);
}
